// BondEmbedding_83786222011211
// MI455X (gfx1250) — matrix operand order repaired and checked by reading, not compiled, not run on hardware
//
#include <hip/hip_runtime.h>

typedef __attribute__((ext_vector_type(16))) _Float16 v16h;
typedef __attribute__((ext_vector_type(8)))  float    v8f;
typedef __attribute__((ext_vector_type(4)))  float    f32x4;

#define IN_DIM  64
#define NG      20
#define K_REAL  (IN_DIM + NG)   // 84
#define KPAD    96              // 3 x K=32 WMMA steps
#define OUT_DIM 128
#define EPB     128             // edges per block
#define LDSTR   104             // padded LDS row stride (halves): 208B -> conflict-free

union Frag16 { uint4 q[2]; v16h v; };
union Pack4  { _Float16 h[4]; uint2 u; };

// One-shot: W [84][128] f32 row-major  ->  W^T f16 [128][96] (k zero-padded) in workspace
__global__ void wt_prep_kernel(const float* __restrict__ W, _Float16* __restrict__ Wt) {
    for (int i = threadIdx.x; i < OUT_DIM * KPAD; i += blockDim.x) {
        int n = i / KPAD;
        int k = i - n * KPAD;
        float v = (k < K_REAL) ? W[k * OUT_DIM + n] : 0.0f;
        Wt[i] = (_Float16)v;
    }
}

__global__ void __launch_bounds__(256)
bond_embed_kernel(const float* __restrict__ bond_feat,
                  const int*   __restrict__ bond_index,
                  const float* __restrict__ pos,
                  const float* __restrict__ bias,
                  const _Float16* __restrict__ Wt,
                  float* __restrict__ out,
                  int E) {
    __shared__ __align__(16) _Float16 sA[EPB][LDSTR];      // edge features, f16
    __shared__ __align__(16) _Float16 sB[OUT_DIM][LDSTR];  // W^T tile, f16

    const int tid = threadIdx.x;
    const int e0  = blockIdx.x * EPB;

    // ---- stage W^T (f16) into LDS, coalesced 16B chunks (KPAD % 8 == 0)
    {
        const uint4* Wt4 = (const uint4*)Wt;
        for (int i = tid; i < OUT_DIM * KPAD / 8; i += 256) {
            int off = i * 8;
            int n = off / KPAD;
            int k = off - n * KPAD;
            *(uint4*)&sB[n][k] = Wt4[i];
        }
    }

    // ---- bond_feat f32 -> f16 into LDS (streaming, non-temporal b128 loads)
    for (int i = tid; i < EPB * (IN_DIM / 4); i += 256) {
        int el = i >> 4;            // edge within block (16 float4 chunks per edge)
        int q  = i & 15;
        int e  = e0 + el;
        Pack4 p;
        if (e < E) {
            f32x4 v = __builtin_nontemporal_load(
                (const f32x4*)bond_feat + (size_t)e * (IN_DIM / 4) + q);
            p.h[0] = (_Float16)v.x; p.h[1] = (_Float16)v.y;
            p.h[2] = (_Float16)v.z; p.h[3] = (_Float16)v.w;
        } else {
            p.u.x = 0u; p.u.y = 0u;
        }
        *(uint2*)&sA[el][q * 4] = p.u;
    }

    // ---- per-edge distance + Gaussian smearing (f16) + zero pad to KPAD
    if (tid < EPB) {
        int e = e0 + tid;
        float d = 0.0f;
        if (e < E) {
            int i0 = bond_index[e];              // bond_index[0][e]
            int i1 = bond_index[(size_t)E + e];  // bond_index[1][e]
            float dx = pos[i0 * 3 + 0] - pos[i1 * 3 + 0];
            float dy = pos[i0 * 3 + 1] - pos[i1 * 3 + 1];
            float dz = pos[i0 * 3 + 2] - pos[i1 * 3 + 2];
            d = sqrtf(dx * dx + dy * dy + dz * dz);
        }
        const float step  = 10.0f / (float)(NG - 1);
        const float coeff = -0.5f / (step * step);
        #pragma unroll
        for (int j = 0; j < NG; ++j) {
            float t = d - step * (float)j;
            float g = (e < E) ? __expf(coeff * t * t) : 0.0f;
            sA[tid][IN_DIM + j] = (_Float16)g;
        }
        #pragma unroll
        for (int k = K_REAL; k < KPAD; ++k) sA[tid][k] = (_Float16)0.0f;
    }
    __syncthreads();

    // ---- WMMA GEMM: each wave = 16 edges x 128 outputs (8 N-tiles, K = 3x32)
    const int lane   = tid & 31;
    const int wave   = tid >> 5;
    const int lo     = lane & 15;
    const bool hiH   = (lane >= 16);

    // A fragments (ISA 16-bit A layout): lanes 0-15 hold K{0..7,16..23},
    // lanes 16-31 hold K{8..15,24..31} of row M = lane&15 -> two 16B LDS reads
    Frag16 a[3];
    {
        int row = wave * 16 + lo;
        int kb  = hiH ? 8 : 0;
        #pragma unroll
        for (int c = 0; c < 3; ++c) {
            a[c].q[0] = *(const uint4*)&sA[row][32 * c + kb];
            a[c].q[1] = *(const uint4*)&sA[row][32 * c + kb + 16];
        }
    }

    // Store addressing: uniform 64-bit base (SGPR) + small per-lane 32-bit offset.
    // Consecutive r rows differ by OUT_DIM floats = 512B -> folds into the
    // store instruction's immediate offset.
    const int rowBase = wave * 16 + (hiH ? 8 : 0);   // local row of first C row
    float* const outBase = out + (size_t)e0 * OUT_DIM;
    const bool fullBlock = (e0 + EPB <= E);          // block-uniform

    #pragma unroll
    for (int nt = 0; nt < 8; ++nt) {
        v8f c = {};
        #pragma unroll
        for (int kc = 0; kc < 3; ++kc) {
            // B fragment (ISA 16-bit B layout): lanes 0-15 hold K{0..7,16..23},
            // lanes 16-31 hold K{8..15,24..31} of column N = lane&15
            Frag16 bf;
            int kb2 = kc * 32 + (hiH ? 8 : 0);
            bf.q[0] = *(const uint4*)&sB[nt * 16 + lo][kb2];
            bf.q[1] = *(const uint4*)&sB[nt * 16 + lo][kb2 + 16];
            c = __builtin_amdgcn_wmma_f32_16x16x32_f16(
                    /*neg_a=*/false, a[kc].v, /*neg_b=*/false, bf.v,
                    /*c_mod=*/(short)0, c, /*reuse_a=*/false, /*reuse_b=*/false);
        }
        // Epilogue: C layout -> lane holds N = lane&15, rows M = r (+8 for hi half)
        float bv = bias[nt * 16 + lo];
        const int voff0 = rowBase * OUT_DIM + nt * 16 + lo;
        if (fullBlock) {
            #pragma unroll
            for (int r = 0; r < 8; ++r)
                __builtin_nontemporal_store(c[r] + bv,
                                            outBase + voff0 + r * OUT_DIM);
        } else {
            #pragma unroll
            for (int r = 0; r < 8; ++r) {
                if (e0 + rowBase + r < E)
                    __builtin_nontemporal_store(c[r] + bv,
                                                outBase + voff0 + r * OUT_DIM);
            }
        }
    }
}

extern "C" void kernel_launch(void* const* d_in, const int* in_sizes, int n_in,
                              void* d_out, int out_size, void* d_ws, size_t ws_size,
                              hipStream_t stream) {
    const float* bond_feat  = (const float*)d_in[0];
    const int*   bond_index = (const int*)d_in[1];
    const float* pos_nodes  = (const float*)d_in[2];
    const float* W          = (const float*)d_in[3];
    const float* b          = (const float*)d_in[4];
    float* out = (float*)d_out;

    const int E = in_sizes[0] / IN_DIM;           // 2,000,000
    _Float16* Wt = (_Float16*)d_ws;               // needs 128*96*2 = 24 KB

    wt_prep_kernel<<<dim3(1), dim3(256), 0, stream>>>(W, Wt);

    const int blocks = (E + EPB - 1) / EPB;       // 15625
    bond_embed_kernel<<<dim3(blocks), dim3(256), 0, stream>>>(
        bond_feat, bond_index, pos_nodes, b, Wt, out, E);
}
